// BiRNNScratch_55198919688224
// MI455X (gfx1250) — matrix operand order repaired and checked by reading, not compiled, not run on hardware
//
#include <hip/hip_runtime.h>
#include <math.h>

typedef __attribute__((ext_vector_type(16))) _Float16 v16h;
typedef __attribute__((ext_vector_type(8)))  _Float16 v8h;
typedef __attribute__((ext_vector_type(8)))  float    v8f;

#define T_   512
#define B_   128
#define D_   256
#define H_   256
#define HSTR 272   // padded LDS row stride (halves): keeps 32B alignment, spreads banks

#if __has_builtin(__builtin_amdgcn_tanhf)
#define TANHF(x) __builtin_amdgcn_tanhf(x)   // v_tanh_f32 (gfx1250 trans op)
#else
#define TANHF(x) tanhf(x)
#endif

// ---------------------------------------------------------------------------
// Prepass: convert the four 256x256 f32 weight matrices to f16 (row-major).
// ---------------------------------------------------------------------------
__global__ void wcvt_kernel(const float* __restrict__ w0, const float* __restrict__ w1,
                            const float* __restrict__ w2, const float* __restrict__ w3,
                            _Float16* __restrict__ o0, _Float16* __restrict__ o1,
                            _Float16* __restrict__ o2, _Float16* __restrict__ o3) {
    int i = blockIdx.x * blockDim.x + threadIdx.x;   // 65536 threads
    o0[i] = (_Float16)w0[i];
    o1[i] = (_Float16)w1[i];
    o2[i] = (_Float16)w2[i];
    o3[i] = (_Float16)w3[i];
}

// ---------------------------------------------------------------------------
// Phase 1: xW[dir] = inputs @ W_xh[dir] + b  (M=T*B=65536, K=D=256, N=H=256)
// One 16x16 output tile per wave; 8 x wmma_f32_16x16x32_f16 over K.
// dir==1 reads time-reversed input rows so xwb[s] == inputs[T-1-s] @ W_xh_b.
// ---------------------------------------------------------------------------
__global__ void __launch_bounds__(256)
xproj_kernel(const float* __restrict__ in,
             const _Float16* __restrict__ wxf, const _Float16* __restrict__ wxb,
             const float* __restrict__ bf,     const float* __restrict__ bb,
             _Float16* __restrict__ xwf,       _Float16* __restrict__ xwb) {
    const int wave = threadIdx.x >> 5;
    const int lane = threadIdx.x & 31;
    const int id   = blockIdx.x * 8 + wave;       // 131072 tiles total
    const int dir  = id >> 16;                    // 65536 tiles per direction
    const int rem  = id & 65535;
    const int mt   = rem >> 4;                    // 4096 M-tiles
    const int nt   = rem & 15;                    // 16  N-tiles

    const int m   = lane & 15;                    // A row / C column index
    const int kb  = (lane >> 4) * 8;              // A K sub-offset per ISA layout
    const int mof = (lane >> 4) * 8;              // C row offset per ISA layout

    const _Float16* wx = dir ? wxb : wxf;
    const float*    bh = dir ? bb  : bf;
    _Float16*       xw = dir ? xwb : xwf;

    const int g    = mt * 16 + m;                                        // dest row
    const int srow = dir ? ((T_ - 1 - (g >> 7)) * B_ + (g & (B_ - 1)))   // reversed time
                         : g;
    const float* arow = in + (size_t)srow * D_ + kb;

    v8f c = {};
#pragma unroll
    for (int ks = 0; ks < 8; ++ks) {
        const float* p = arow + ks * 32;
        v8h lo, hi;
#pragma unroll
        for (int i = 0; i < 8; ++i) { lo[i] = (_Float16)p[i]; hi[i] = (_Float16)p[16 + i]; }
        v16h a = __builtin_shufflevector(lo, hi, 0,1,2,3,4,5,6,7,8,9,10,11,12,13,14,15);
        // B 32x16: lane%16 = N column, elements = K rows in A's order (kb+i, 16+kb+i)
        v16h b;
#pragma unroll
        for (int i = 0; i < 8; ++i) {
            b[i]     = wx[(size_t)(ks * 32 + kb + i) * H_ + nt * 16 + m];
            b[8 + i] = wx[(size_t)(ks * 32 + 16 + kb + i) * H_ + nt * 16 + m];
        }
        c = __builtin_amdgcn_wmma_f32_16x16x32_f16(false, a, false, b, (short)0, c, false, false);
    }

    const float bias = bh[nt * 16 + m];
    _Float16* orow = xw + (size_t)(mt * 16) * H_ + nt * 16 + m;
#pragma unroll
    for (int i = 0; i < 8; ++i)
        orow[(size_t)(i + mof) * H_] = (_Float16)(c[i] + bias);
}

// ---------------------------------------------------------------------------
// Phase 2: persistent scan. 16 blocks = 2 dirs x 8 batch-tiles (16 rows each).
// W_hh B-tiles live in REGISTERS (loop-invariant, 64 VGPRs/wave); only the
// 16x256 h slice lives in LDS, double-buffered (one barrier per step).
// xW accumulator-init values are software-pipelined (prefetch t+1 during t).
// ---------------------------------------------------------------------------
__global__ void __launch_bounds__(512)
scan_kernel(const _Float16* __restrict__ xwf, const _Float16* __restrict__ xwb,
            const _Float16* __restrict__ whf, const _Float16* __restrict__ whb,
            float* __restrict__ out) {
    __shared__ _Float16 lds_h[2][16 * HSTR];   // double-buffered h slice (~17 KB)

    const int dir  = blockIdx.x >> 3;
    const int bt   = blockIdx.x & 7;
    const int nt   = threadIdx.x >> 5;         // wave id = N tile (16 waves)
    const int lane = threadIdx.x & 31;
    const int m    = lane & 15;
    const int kb   = (lane >> 4) * 8;
    const int mof  = (lane >> 4) * 8;

    const _Float16* wh = dir ? whb : whf;
    const _Float16* xw = dir ? xwb : xwf;

    // Hoist the 8 loop-invariant W_hh B-tiles into registers (B 32x16 layout:
    // lane%16 = N column, elements = K rows in A's order (kb+i, 16+kb+i)).
    v16h breg[8];
#pragma unroll
    for (int ks = 0; ks < 8; ++ks)
#pragma unroll
        for (int i = 0; i < 8; ++i) {
            breg[ks][i]     = wh[(size_t)(ks * 32 + kb + i) * H_ + nt * 16 + m];
            breg[ks][8 + i] = wh[(size_t)(ks * 32 + 16 + kb + i) * H_ + nt * 16 + m];
        }

    // h0 = 0 in buffer 0.
    for (int i = threadIdx.x; i < 16 * HSTR; i += 512)
        lds_h[0][i] = (_Float16)0.f;
    __syncthreads();

    const int g0 = bt * 16;                                           // batch row base
    float* fin = out + (size_t)T_ * B_ * (2 * H_) + (size_t)dir * B_ * H_;

    // Prefetch xW for t = 0.
    _Float16 xr[8];
    {
        const _Float16* x0 = xw + (size_t)g0 * H_ + nt * 16 + m;
#pragma unroll
        for (int i = 0; i < 8; ++i) xr[i] = x0[(size_t)(i + mof) * H_];
    }

    for (int t = 0; t < T_; ++t) {
        const _Float16* hcur = lds_h[t & 1];
        _Float16*       hnxt = lds_h[(t + 1) & 1];

        v8f c;
#pragma unroll
        for (int i = 0; i < 8; ++i) c[i] = (float)xr[i];

        // Prefetch xW for t+1; latency hides under the WMMA chain below.
        if (t + 1 < T_) {
            const _Float16* xn = xw + ((size_t)(t + 1) * B_ + g0) * H_ + nt * 16 + m;
#pragma unroll
            for (int i = 0; i < 8; ++i) xr[i] = xn[(size_t)(i + mof) * H_];
        }

#pragma unroll
        for (int ks = 0; ks < 8; ++ks) {
            const _Float16* ap = hcur + m * HSTR + ks * 32 + kb;
            v8h lo = *(const v8h*)ap;
            v8h hi = *(const v8h*)(ap + 16);
            v16h a = __builtin_shufflevector(lo, hi, 0,1,2,3,4,5,6,7,8,9,10,11,12,13,14,15);
            c = __builtin_amdgcn_wmma_f32_16x16x32_f16(false, a, false, breg[ks],
                                                       (short)0, c, false, false);
        }
#pragma unroll
        for (int i = 0; i < 8; ++i) c[i] = TANHF(c[i]);

        // Write h_{t+1} into the other buffer (no race with hcur readers) and
        // stream outputs; one barrier publishes hnxt for the next step.
        const int tp = dir ? (T_ - 1 - t) : t;
        float* orow = out + ((size_t)tp * B_ + g0) * (2 * H_) + dir * H_ + nt * 16 + m;
#pragma unroll
        for (int i = 0; i < 8; ++i) {
            const int M = i + mof;
            hnxt[M * HSTR + nt * 16 + m] = (_Float16)c[i];
            orow[(size_t)M * (2 * H_)] = c[i];
        }
        if (t == T_ - 1) {                                            // final states
            float* frow = fin + (size_t)g0 * H_ + nt * 16 + m;
#pragma unroll
            for (int i = 0; i < 8; ++i) frow[(size_t)(i + mof) * H_] = c[i];
        }
        __syncthreads();
    }
}

// ---------------------------------------------------------------------------
extern "C" void kernel_launch(void* const* d_in, const int* in_sizes, int n_in,
                              void* d_out, int out_size, void* d_ws, size_t ws_size,
                              hipStream_t stream) {
    const float* in  = (const float*)d_in[0];
    const float* Wxf = (const float*)d_in[1];
    const float* Whf = (const float*)d_in[2];
    const float* bf  = (const float*)d_in[3];
    const float* Wxb = (const float*)d_in[4];
    const float* Whb = (const float*)d_in[5];
    const float* bb  = (const float*)d_in[6];
    float* out = (float*)d_out;

    // Workspace: 4 x 128KB f16 weights, then 2 x 33.5MB f16 xW  (~67.6 MB)
    _Float16* wxf16 = (_Float16*)d_ws;
    _Float16* whf16 = wxf16 + 65536;
    _Float16* wxb16 = whf16 + 65536;
    _Float16* whb16 = wxb16 + 65536;
    _Float16* xwf   = whb16 + 65536;
    _Float16* xwb   = xwf + (size_t)T_ * B_ * H_;

    wcvt_kernel<<<256, 256, 0, stream>>>(Wxf, Whf, Wxb, Whb, wxf16, whf16, wxb16, whb16);
    xproj_kernel<<<16384, 256, 0, stream>>>(in, wxf16, wxb16, bf, bb, xwf, xwb);
    scan_kernel<<<16, 512, 0, stream>>>(xwf, xwb, whf16, whb16, out);
}
